// UNIN_59536836657978
// MI455X (gfx1250) — matrix operand order repaired and checked by reading, not compiled, not run on hardware
//
#include <hip/hip_runtime.h>
#include <hip/hip_bf16.h>

typedef __attribute__((ext_vector_type(16))) _Float16 v16h;
typedef __attribute__((ext_vector_type(8)))  _Float16 v8h;
typedef __attribute__((ext_vector_type(4)))  _Float16 v4h;
typedef __attribute__((ext_vector_type(8)))  float    v8f;

#define KN      50000
#define KE      800000
#define KETOT   (KE + KN)
#define KHID    64
#define KFEA    16
#define KPAD    160   // max padded K (144 -> 160)

// ---------------------------------------------------------------------------
// Weight packing: f32 W[out_d][in_d] -> WMMA B-fragment layout, f16.
// B frag layout (16x16x32 f16, wave32): dst[((kc*NT+nt)*32 + lane)*16 + h]
//   lane<16 : K = kc*32 + h,      N = nt*16 + lane
//   lane>=16: K = kc*32 + 16 + h, N = nt*16 + (lane-16)
// ---------------------------------------------------------------------------
__global__ void pack_b_kernel(const float* __restrict__ W, _Float16* __restrict__ dst,
                              int out_d, int in_d, int kchunks) {
    int NT = out_d >> 4;
    int total = kchunks * NT * 32 * 16;
    for (int i = blockIdx.x * blockDim.x + threadIdx.x; i < total;
         i += gridDim.x * blockDim.x) {
        int h    = i & 15;
        int lane = (i >> 4) & 31;
        int t    = i >> 9;
        int nt   = t % NT;
        int kc   = t / NT;
        int k = kc * 32 + ((lane >= 16) ? 16 : 0) + h;
        int n = nt * 16 + (lane & 15);
        float v = (k < in_d) ? W[(size_t)n * in_d + k] : 0.0f;
        dst[i] = (_Float16)v;
    }
}

// ---------------------------------------------------------------------------
// Dual-output WMMA GEMM:  out = concat(p0,p1,p2)(N x K) @ Wt(K x out_d) + bias
// Block = 256 threads = 8 waves; each wave owns one 16-row tile.
// A staged in LDS as f16 (per-wave private region -> wave-synchronous, no
// barriers; LDS ops are in-order per wave).
// ---------------------------------------------------------------------------
template <int NT, bool DUAL>
__global__ void gemm_wmma_kernel(const float* __restrict__ p0, int w0,
                                 const float* __restrict__ p1, int w1,
                                 const float* __restrict__ p2, int w2,
                                 const _Float16* __restrict__ bfA,
                                 const float* __restrict__ biasA,
                                 float* __restrict__ outA,
                                 const _Float16* __restrict__ bfB,
                                 const float* __restrict__ biasB,
                                 float* __restrict__ outB,
                                 int nrows, int kchunks) {
    __shared__ alignas(16) _Float16 lds[8 * 16 * KPAD];

    const int wave = threadIdx.x >> 5;
    const int lane = threadIdx.x & 31;
    const int tile = blockIdx.x * 8 + wave;
    const int rowbase = tile * 16;
    if (rowbase >= nrows) return;                 // no __syncthreads used

    _Float16* ldsT = lds + wave * 16 * KPAD;

    const int K    = w0 + w1 + w2;
    const int padK = kchunks * 32;

    // zero the K..padK padding region of this wave's tile
    const int pcnt = padK - K;
    if (pcnt > 0) {
        for (int idx = lane; idx < 16 * pcnt; idx += 32) {
            int row = idx / pcnt;
            int kk  = K + (idx - row * pcnt);
            ldsT[row * KPAD + kk] = (_Float16)0.0f;
        }
    }

    // stage A tile: float4 loads from the 3 concat sources -> f16 in LDS
    const int nv4 = K >> 2;
    for (int idx = lane; idx < 16 * nv4; idx += 32) {
        int row = idx / nv4;
        int j   = (idx - row * nv4) << 2;         // column (multiple of 4)
        int grow = rowbase + row;
        float4 f;
        if (j < w0) {
            f = ((const float4*)(p0 + (size_t)grow * w0))[j >> 2];
        } else if (j < w0 + w1) {
            f = ((const float4*)(p1 + (size_t)grow * w1))[(j - w0) >> 2];
        } else {
            f = ((const float4*)(p2 + (size_t)grow * w2))[(j - w0 - w1) >> 2];
        }
        v4h h4 = { (_Float16)f.x, (_Float16)f.y, (_Float16)f.z, (_Float16)f.w };
        *(v4h*)(ldsT + row * KPAD + j) = h4;
    }

    v8f accA[NT];
    v8f accB[NT];
#pragma unroll
    for (int nt = 0; nt < NT; ++nt) { accA[nt] = (v8f)0.0f; accB[nt] = (v8f)0.0f; }

    const int arow  = lane & 15;
    const int koffs = ((lane >> 4) & 1) * 16;

    for (int kc = 0; kc < kchunks; ++kc) {
        const int kbase = kc * 32 + koffs;
        // A fragment: halves 0..15 = K kbase..kbase+15
        v8h lo = *(const v8h*)(ldsT + arow * KPAD + kbase);
        v8h hi = *(const v8h*)(ldsT + arow * KPAD + kbase + 8);
        v16h a = __builtin_shufflevector(lo, hi, 0, 1, 2, 3, 4, 5, 6, 7,
                                                 8, 9, 10, 11, 12, 13, 14, 15);
#pragma unroll
        for (int nt = 0; nt < NT; ++nt) {
            size_t bofs = ((size_t)(kc * NT + nt) * 32 + lane) * 16;
            v16h b = *(const v16h*)(bfA + bofs);
            accA[nt] = __builtin_amdgcn_wmma_f32_16x16x32_f16(
                false, a, false, b, (short)0, accA[nt], false, false);
            if (DUAL) {
                v16h b2 = *(const v16h*)(bfB + bofs);
                accB[nt] = __builtin_amdgcn_wmma_f32_16x16x32_f16(
                    false, a, false, b2, (short)0, accB[nt], false, false);
            }
        }
    }

    // D layout: lane<16 -> col=lane, M=r ; lane>=16 -> col=lane-16, M=r+8
    const int col   = lane & 15;
    const int mbase = (lane >> 4) * 8;
    const int out_d = NT * 16;
#pragma unroll
    for (int nt = 0; nt < NT; ++nt) {
        int gc = nt * 16 + col;
        float ba = biasA[gc];
#pragma unroll
        for (int r = 0; r < 8; ++r) {
            int m = mbase + r;
            outA[(size_t)(rowbase + m) * out_d + gc] = accA[nt][r] + ba;
        }
        if (DUAL) {
            float bb = biasB[gc];
#pragma unroll
            for (int r = 0; r < 8; ++r) {
                int m = mbase + r;
                outB[(size_t)(rowbase + m) * out_d + gc] = accB[nt][r] + bb;
            }
        }
    }
}

// ---------------------------------------------------------------------------
// Edge-phase kernels (memory bound; f32 with float4 gathers + f32 atomics)
// ---------------------------------------------------------------------------
__device__ __forceinline__ void atomicMaxF(float* a, float v) {
    if (v >= 0.0f) atomicMax((int*)a, __float_as_int(v));
    else           atomicMin((unsigned int*)a, __float_as_uint(v));
}

template <int OD>
__global__ void init_gat_kernel(float* __restrict__ out, const float* __restrict__ bias,
                                float* __restrict__ emax, float* __restrict__ denom) {
    int tid = blockIdx.x * blockDim.x + threadIdx.x;
    if (tid >= KN * OD) return;
    out[tid] = bias[tid % OD];          // segment_sum lands on top of bias
    if ((tid % OD) == 0) {
        emax[tid / OD]  = -3.0e38f;
        denom[tid / OD] = 0.0f;
    }
}

template <int OD>
__global__ void edge_e_kernel(const long long* __restrict__ ei,
                              const float* __restrict__ xl,
                              const float* __restrict__ xr,
                              const float* __restrict__ att,
                              float* __restrict__ e_buf,
                              float* __restrict__ emax) {
    int e = blockIdx.x * blockDim.x + threadIdx.x;
    if (e >= KETOT) return;
    long long s, d;
    if (e < KE) { s = ei[e]; d = ei[KE + e]; } else { s = d = (long long)(e - KE); }
    const float4* xls = (const float4*)(xl + (size_t)s * OD);
    const float4* xrd = (const float4*)(xr + (size_t)d * OD);
    const float4* at4 = (const float4*)att;
    float acc = 0.0f;
#pragma unroll
    for (int j = 0; j < OD / 4; ++j) {
        float4 a = xls[j], b = xrd[j], w = at4[j];
        float v0 = a.x + b.x; v0 = v0 > 0.0f ? v0 : 0.2f * v0;
        float v1 = a.y + b.y; v1 = v1 > 0.0f ? v1 : 0.2f * v1;
        float v2 = a.z + b.z; v2 = v2 > 0.0f ? v2 : 0.2f * v2;
        float v3 = a.w + b.w; v3 = v3 > 0.0f ? v3 : 0.2f * v3;
        acc += v0 * w.x + v1 * w.y + v2 * w.z + v3 * w.w;
    }
    e_buf[e] = acc;
    atomicMaxF(&emax[d], acc);
}

__global__ void edge_a_kernel(const long long* __restrict__ ei,
                              const float* __restrict__ e_buf,
                              const float* __restrict__ emax,
                              float* __restrict__ a_buf,
                              float* __restrict__ denom) {
    int e = blockIdx.x * blockDim.x + threadIdx.x;
    if (e >= KETOT) return;
    long long d = (e < KE) ? ei[KE + e] : (long long)(e - KE);
    float a = __expf(e_buf[e] - emax[d]);
    a_buf[e] = a;
    atomicAdd(&denom[d], a);
}

template <int OD>
__global__ void edge_scatter_kernel(const long long* __restrict__ ei,
                                    const float* __restrict__ xl,
                                    const float* __restrict__ a_buf,
                                    const float* __restrict__ denom,
                                    float* __restrict__ out) {
    const int PARTS = OD / 4;
    long long tid = (long long)blockIdx.x * blockDim.x + threadIdx.x;
    if (tid >= (long long)KETOT * PARTS) return;
    int e    = (int)(tid / PARTS);
    int part = (int)(tid - (long long)e * PARTS);
    long long s, d;
    if (e < KE) { s = ei[e]; d = ei[KE + e]; } else { s = d = (long long)(e - KE); }
    float alpha = a_buf[e] / denom[d];
    float4 v = ((const float4*)(xl + (size_t)s * OD))[part];
    float* o = out + (size_t)d * OD + part * 4;
    atomicAdd(o + 0, alpha * v.x);
    atomicAdd(o + 1, alpha * v.y);
    atomicAdd(o + 2, alpha * v.z);
    atomicAdd(o + 3, alpha * v.w);
}

// ---------------------------------------------------------------------------
// Elementwise helpers
// ---------------------------------------------------------------------------
__global__ void static_kernel(const int* __restrict__ mask,
                              const float* __restrict__ W, const float* __restrict__ b,
                              float* __restrict__ st) {
    int tid = blockIdx.x * blockDim.x + threadIdx.x;
    if (tid >= KN * KHID) return;
    int n = tid >> 6, j = tid & 63;
    float f0 = (mask[n] != 0) ? 1.0f : 0.0f;
    float f1 = 1.0f - f0;
    float v = W[j * 2] * f0 + W[j * 2 + 1] * f1 + b[j];
    st[tid] = v > 0.0f ? v : 0.0f;
}

__global__ void mask_x_kernel(const int* __restrict__ mask, const float* __restrict__ x,
                              float* __restrict__ xm) {
    int tid = blockIdx.x * blockDim.x + threadIdx.x;
    if (tid >= KN * KFEA) return;
    int n = tid / KFEA;
    xm[tid] = (mask[n] != 0) ? x[tid] : 0.0f;
}

__global__ void x2_kernel(const int* __restrict__ mask, const float* __restrict__ xm,
                          const float* __restrict__ y2, float* __restrict__ x2) {
    int tid = blockIdx.x * blockDim.x + threadIdx.x;
    if (tid >= KN * KFEA) return;
    int n = tid / KFEA;
    x2[tid] = (mask[n] != 0) ? xm[tid] : y2[tid];
}

// graw1 := sigmoid(graw1)*hidden (=reset*h), graw2 := sigmoid(graw2) (=update)
__global__ void gate_kernel(float* __restrict__ graw1, float* __restrict__ graw2,
                            const float* __restrict__ hidden) {
    int tid = blockIdx.x * blockDim.x + threadIdx.x;
    if (tid >= KN * KHID) return;
    float r = 1.0f / (1.0f + __expf(-graw1[tid]));
    float u = 1.0f / (1.0f + __expf(-graw2[tid]));
    graw1[tid] = r * hidden[tid];
    graw2[tid] = u;
}

// h := u*h + (1-u)*tanh(craw)
__global__ void hidden_kernel(float* __restrict__ hidden, const float* __restrict__ u,
                              const float* __restrict__ craw) {
    int tid = blockIdx.x * blockDim.x + threadIdx.x;
    if (tid >= KN * KHID) return;
    float c = tanhf(craw[tid]);
    float uu = u[tid];
    hidden[tid] = uu * hidden[tid] + (1.0f - uu) * c;
}

__global__ void relu_kernel(const float* __restrict__ in, float* __restrict__ out) {
    int tid = blockIdx.x * blockDim.x + threadIdx.x;
    if (tid >= KN * KHID) return;
    float v = in[tid];
    out[tid] = v > 0.0f ? v : 0.0f;
}

// ---------------------------------------------------------------------------
// Host orchestration
// ---------------------------------------------------------------------------
extern "C" void kernel_launch(void* const* d_in, const int* in_sizes, int n_in,
                              void* d_out, int out_size, void* d_ws, size_t ws_size,
                              hipStream_t stream) {
    (void)in_sizes; (void)n_in; (void)out_size; (void)ws_size;

    const float*     x      = (const float*)d_in[0];
    const long long* ei     = (const long long*)d_in[1];
    const int*       mask   = (const int*)d_in[2];
    const float*     stW    = (const float*)d_in[6];
    const float*     stB    = (const float*)d_in[7];
    struct Gat { const float *Wl, *bl, *Wr, *br, *att, *bias; };
    auto getg = [&](int base) {
        Gat p; p.Wl = (const float*)d_in[base];     p.bl = (const float*)d_in[base + 1];
               p.Wr = (const float*)d_in[base + 2]; p.br = (const float*)d_in[base + 3];
               p.att = (const float*)d_in[base + 4]; p.bias = (const float*)d_in[base + 5];
        return p;
    };
    Gat fillp = getg(8), resetp = getg(14), updatep = getg(20), cellp = getg(26), finp = getg(32);
    const float* predW = (const float*)d_in[38];
    const float* predB = (const float*)d_in[39];
    float* out = (float*)d_out;

    // ----- workspace carve-up -----
    char* ws = (char*)d_ws; size_t off = 0;
    auto carve = [&](size_t bytes) { void* p = ws + off; off += (bytes + 255) & ~(size_t)255; return p; };
    float* st      = (float*)carve((size_t)KN * KHID * 4);
    float* hidden  = (float*)carve((size_t)KN * KHID * 4);
    float* xm      = (float*)carve((size_t)KN * KFEA * 4);
    float* y2      = (float*)carve((size_t)KN * KFEA * 4);
    float* x2      = (float*)carve((size_t)KN * KFEA * 4);
    float* xlb     = (float*)carve((size_t)KN * KHID * 4);
    float* xrb     = (float*)carve((size_t)KN * KHID * 4);
    float* graw1   = (float*)carve((size_t)KN * KHID * 4);   // reset raw -> rh -> s
    float* graw2   = (float*)carve((size_t)KN * KHID * 4);   // update raw -> update
    float* craw    = (float*)carve((size_t)KN * KHID * 4);   // cell raw -> final raw
    float* e_buf   = (float*)carve((size_t)KETOT * 4);
    float* a_buf   = (float*)carve((size_t)KETOT * 4);
    float* emax    = (float*)carve((size_t)KN * 4);
    float* denom   = (float*)carve((size_t)KN * 4);
    const size_t BF = 5 * 4 * 32 * 16;                       // max frag halves
    _Float16* bfrag[11];
    for (int i = 0; i < 11; ++i) bfrag[i] = (_Float16*)carve(BF * 2);

    // ----- pack weights into WMMA B fragments -----
    auto pack = [&](const float* W, _Float16* dst, int od, int id, int kc) {
        int total = kc * (od >> 4) * 32 * 16;
        pack_b_kernel<<<(total + 255) / 256, 256, 0, stream>>>(W, dst, od, id, kc);
    };
    pack(fillp.Wl,   bfrag[0], 16, 144, 5);  pack(fillp.Wr,   bfrag[1], 16, 144, 5);
    pack(resetp.Wl,  bfrag[2], 64, 144, 5);  pack(resetp.Wr,  bfrag[3], 64, 144, 5);
    pack(updatep.Wl, bfrag[4], 64, 144, 5);  pack(updatep.Wr, bfrag[5], 64, 144, 5);
    pack(cellp.Wl,   bfrag[6], 64, 144, 5);  pack(cellp.Wr,   bfrag[7], 64, 144, 5);
    pack(finp.Wl,    bfrag[8], 64, 144, 5);  pack(finp.Wr,    bfrag[9], 64, 144, 5);
    pack(predW,      bfrag[10], 16, 128, 4);

    hipMemsetAsync(hidden, 0, (size_t)KN * KHID * 4, stream);
    static_kernel<<<(KN * KHID + 255) / 256, 256, 0, stream>>>(mask, stW, stB, st);

    const int gemmBlocks = ((KN / 16) + 7) / 8;              // 3125 tiles / 8 waves

    auto run_gat = [&](const Gat& g, _Float16* bL, _Float16* bR,
                       const float* p2, float* gout, int od) {
        if (od == 16) {
            gemm_wmma_kernel<1, true><<<gemmBlocks, 256, 0, stream>>>(
                xm, 16, st, 64, p2, 64, bL, g.bl, xlb, bR, g.br, xrb, KN, 5);
        } else {
            gemm_wmma_kernel<4, true><<<gemmBlocks, 256, 0, stream>>>(
                x2, 16, st, 64, p2, 64, bL, g.bl, xlb, bR, g.br, xrb, KN, 5);
        }
        int eb = (KETOT + 255) / 256;
        if (od == 16) {
            init_gat_kernel<16><<<(KN * 16 + 255) / 256, 256, 0, stream>>>(gout, g.bias, emax, denom);
            edge_e_kernel<16><<<eb, 256, 0, stream>>>(ei, xlb, xrb, g.att, e_buf, emax);
            edge_a_kernel<<<eb, 256, 0, stream>>>(ei, e_buf, emax, a_buf, denom);
            long long sc = (long long)KETOT * 4;
            edge_scatter_kernel<16><<<(unsigned)((sc + 255) / 256), 256, 0, stream>>>(ei, xlb, a_buf, denom, gout);
        } else {
            init_gat_kernel<64><<<(KN * 64 + 255) / 256, 256, 0, stream>>>(gout, g.bias, emax, denom);
            edge_e_kernel<64><<<eb, 256, 0, stream>>>(ei, xlb, xrb, g.att, e_buf, emax);
            edge_a_kernel<<<eb, 256, 0, stream>>>(ei, e_buf, emax, a_buf, denom);
            long long sc = (long long)KETOT * 16;
            edge_scatter_kernel<64><<<(unsigned)((sc + 255) / 256), 256, 0, stream>>>(ei, xlb, a_buf, denom, gout);
        }
    };

    const int nb16 = (KN * KFEA + 255) / 256;
    const int nb64 = (KN * KHID + 255) / 256;

    for (int t = 0; t < 10; ++t) {
        const float* xt = x + (size_t)t * KN * KFEA;
        mask_x_kernel<<<nb16, 256, 0, stream>>>(mask, xt, xm);

        // fill GAT: [xm(16), static(64), hidden(64)] -> y2 (N x 16)
        run_gat(fillp, bfrag[0], bfrag[1], hidden, y2, 16);
        x2_kernel<<<nb16, 256, 0, stream>>>(mask, xm, y2, x2);

        // reset / update GATs on [x2, static, hidden]
        run_gat(resetp,  bfrag[2], bfrag[3], hidden, graw1, 64);
        run_gat(updatep, bfrag[4], bfrag[5], hidden, graw2, 64);
        gate_kernel<<<nb64, 256, 0, stream>>>(graw1, graw2, hidden);   // graw1=rh, graw2=u

        // cell GAT on [x2, static, rh]
        run_gat(cellp, bfrag[6], bfrag[7], graw1, craw, 64);
        hidden_kernel<<<nb64, 256, 0, stream>>>(hidden, graw2, craw);

        // final GAT on [x2, static, hidden_new]
        run_gat(finp, bfrag[8], bfrag[9], hidden, craw, 64);
        relu_kernel<<<nb64, 256, 0, stream>>>(craw, graw1);            // s

        // pred: [s(64), static(64)] @ predW.T + predB -> d_out[t]
        gemm_wmma_kernel<1, false><<<gemmBlocks, 256, 0, stream>>>(
            graw1, 64, st, 64, (const float*)nullptr, 0,
            bfrag[10], predB, out + (size_t)t * KN * KFEA,
            (const _Float16*)nullptr, (const float*)nullptr, (float*)nullptr,
            KN, 4);
    }
}
